// BipartiteGNN_78469052498162
// MI455X (gfx1250) — matrix operand order repaired and checked by reading, not compiled, not run on hardware
//
#include <hip/hip_runtime.h>

typedef __attribute__((ext_vector_type(16))) _Float16 v16h;
typedef __attribute__((ext_vector_type(8)))  float    v8f;

#define HDIM 64
#define WPB  8   // waves per block (blockDim = 256)

// ---------------------------------------------------------------------------
// Fragment layout helpers (CDNA5 wave32 WMMA 16x16x32 f16):
//  A (16x32, MxK): lane L holds row M=L&15.
//    lanes 0-15 : i<8 -> K=i,      i>=8 -> K=i+8   ({0..7,16..23})
//    lanes 16-31: i<8 -> K=i+8,    i>=8 -> K=i+16  ({8..15,24..31})
//  B (32x16, KxN): lane L holds col N=L&15.
//    lanes 0-15 : i<8 -> K=i,      i>=8 -> K=i+8   ({0..7,16..23})
//    lanes 16-31: i<8 -> K=i+8,    i>=8 -> K=i+16  ({8..15,24..31})
//  C/D (16x16 f32): lane L holds col N=L&15; vgpr d holds row M=d (+8 if L>=16)
// ---------------------------------------------------------------------------

__device__ __forceinline__ v16h load_a_f32(const float* __restrict__ row, int kbase, int lane) {
  int off = kbase + ((lane & 16) ? 8 : 0);
  v16h a;
#pragma unroll
  for (int i = 0; i < 8; ++i) a[i] = (_Float16)row[off + i];
#pragma unroll
  for (int i = 0; i < 8; ++i) a[i + 8] = (_Float16)row[off + 16 + i];
  return a;
}

__device__ __forceinline__ v16h load_a_f16(const _Float16* __restrict__ row, int kbase, int lane) {
  int off = kbase + ((lane & 16) ? 8 : 0);
  v16h a;
#pragma unroll
  for (int i = 0; i < 8; ++i) a[i] = row[off + i];
#pragma unroll
  for (int i = 0; i < 8; ++i) a[i + 8] = row[off + 16 + i];
  return a;
}

// Pack row-major W[K x 64] (f32) into B-fragment order (f16).
// out layout: [kc][nt=4][lane=32][i=16]; grid.x = K/32, block = 128.
__global__ void pack_w_kernel(const float* __restrict__ W, _Float16* __restrict__ out) {
  int kc   = blockIdx.x;
  int lane = threadIdx.x & 31;
  int nt   = threadIdx.x >> 5;
  int n     = nt * 16 + (lane & 15);
  int kbase = kc * 32 + ((lane & 16) ? 8 : 0);
  _Float16* dst = out + (((kc * 4 + nt) * 32 + lane) << 4);
#pragma unroll
  for (int i = 0; i < 16; ++i)
    dst[i] = (_Float16)W[(kbase + i + ((i >= 8) ? 8 : 0)) * HDIM + n];
}

// Small-K input projection: out[N,64] = x[N,K] @ w[K,64] + b
__global__ void input_proj_kernel(const float* __restrict__ x, int K,
                                  const float* __restrict__ w, const float* __restrict__ b,
                                  float* __restrict__ out, long total) {
  long t = (long)blockIdx.x * blockDim.x + threadIdx.x;
  if (t >= total) return;
  long row = t >> 6;
  int  col = (int)(t & 63);
  float s = b[col];
  for (int k = 0; k < K; ++k) s += x[row * K + k] * w[k * HDIM + col];
  out[t] = s;
}

// msg = sum / max(count, 1)
__global__ void scatter_fin_kernel(float* __restrict__ sum, const float* __restrict__ cnt, long total) {
  long t = (long)blockIdx.x * blockDim.x + threadIdx.x;
  if (t >= total) return;
  sum[t] = sum[t] / fmaxf(cnt[t >> 6], 1.0f);
}

// ---------------------------------------------------------------------------
// Fused edge message MLP + scatter-add:
//   x_e = [featA[idxA[e]](64) | featB[idxB[e]](64) | ea[e](1)]  (129)
//   h   = relu(x_e @ W1 + b1)  ;  m = h @ W2 + b2
//   atomic: sum[idxA[e]] += m ; cnt[idxA[e]] += 1
// ---------------------------------------------------------------------------
__global__ void __launch_bounds__(256) edge_msg_kernel(
    const float* __restrict__ featA, const float* __restrict__ featB,
    const int* __restrict__ idxA, const int* __restrict__ idxB,
    const float* __restrict__ ea, int E,
    const _Float16* __restrict__ pW1, const float* __restrict__ w1last,
    const float* __restrict__ b1,
    const _Float16* __restrict__ pW2, const float* __restrict__ b2,
    float* __restrict__ sumOut, float* __restrict__ cntOut)
{
  __shared__ _Float16 hbuf[WPB][16][HDIM];
  int wave = threadIdx.x >> 5;
  int lane = threadIdx.x & 31;
  int tile = blockIdx.x * WPB + wave;
  if (tile * 16 >= E) return;
  int rowsValid = E - tile * 16;
  if (rowsValid > 16) rowsValid = 16;

  int r = lane & 15;
  int e = tile * 16 + r;
  if (e >= E) e = E - 1;
  int   ia  = idxA[e];
  int   ib  = idxB[e];
  float eav = ea[e];
  const float* rowA = featA + (long)ia * HDIM;
  const float* rowB = featB + (long)ib * HDIM;

  // ---- GEMM1: 16x129 @ 129x64 (K=128 via WMMA, last row via VALU) ----
  v8f acc[4] = {};
#pragma unroll
  for (int kc = 0; kc < 4; ++kc) {
    const float* rp = (kc < 2) ? rowA : rowB;
    v16h a = load_a_f32(rp, (kc & 1) * 32, lane);
#pragma unroll
    for (int nt = 0; nt < 4; ++nt) {
      v16h b = *(const v16h*)(pW1 + (((kc * 4 + nt) * 32 + lane) << 4));
      acc[nt] = __builtin_amdgcn_wmma_f32_16x16x32_f16(false, a, false, b,
                                                       (short)0, acc[nt], false, false);
    }
  }

  int nc   = lane & 15;
  int mofs = (lane & 16) ? 8 : 0;
#pragma unroll
  for (int nt = 0; nt < 4; ++nt) {
    int   n  = nt * 16 + nc;
    float wl = w1last[n];
    float bb = b1[n];
#pragma unroll
    for (int d = 0; d < 8; ++d) {
      int   m  = d + mofs;
      float em = __shfl(eav, m, 32);
      float v  = acc[nt][d] + em * wl + bb;
      hbuf[wave][m][n] = (_Float16)fmaxf(v, 0.0f);
    }
  }
  __threadfence_block();

  // ---- GEMM2: 16x64 @ 64x64 ----
  v8f acc2[4] = {};
#pragma unroll
  for (int kc = 0; kc < 2; ++kc) {
    v16h a = load_a_f16(&hbuf[wave][r][0], kc * 32, lane);
#pragma unroll
    for (int nt = 0; nt < 4; ++nt) {
      v16h b = *(const v16h*)(pW2 + (((kc * 4 + nt) * 32 + lane) << 4));
      acc2[nt] = __builtin_amdgcn_wmma_f32_16x16x32_f16(false, a, false, b,
                                                        (short)0, acc2[nt], false, false);
    }
  }

  // ---- scatter-add ----
#pragma unroll
  for (int nt = 0; nt < 4; ++nt) {
    int   n  = nt * 16 + nc;
    float bb = b2[n];
#pragma unroll
    for (int d = 0; d < 8; ++d) {
      int   m   = d + mofs;
      int   si  = __shfl(ia, m, 32);
      float val = acc2[nt][d] + bb;
      if (m < rowsValid) atomicAdd(&sumOut[(long)si * HDIM + n], val);
    }
  }
  if (lane < 16 && r < rowsValid) atomicAdd(&cntOut[ia], 1.0f);
}

// ---------------------------------------------------------------------------
// Node MLP: out = ( relu(LN([x|msg] @ L1 + b1)) ) @ L2 + b2
// ---------------------------------------------------------------------------
__global__ void __launch_bounds__(256) node_mlp_kernel(
    const float* __restrict__ x, const float* __restrict__ msg,
    const _Float16* __restrict__ pL1, const float* __restrict__ b1,
    const float* __restrict__ ln_g, const float* __restrict__ ln_b,
    const _Float16* __restrict__ pL2, const float* __restrict__ b2,
    float* __restrict__ out, int N)
{
  __shared__ _Float16 hbuf[WPB][16][HDIM];
  int wave = threadIdx.x >> 5;
  int lane = threadIdx.x & 31;
  int tile = blockIdx.x * WPB + wave;
  if (tile * 16 >= N) return;
  int rowsValid = N - tile * 16;
  if (rowsValid > 16) rowsValid = 16;

  int r = lane & 15;
  long node = (long)tile * 16 + r;
  if (node >= N) node = N - 1;
  const float* rowA = x   + node * HDIM;
  const float* rowB = msg + node * HDIM;

  v8f acc[4] = {};
#pragma unroll
  for (int kc = 0; kc < 4; ++kc) {
    const float* rp = (kc < 2) ? rowA : rowB;
    v16h a = load_a_f32(rp, (kc & 1) * 32, lane);
#pragma unroll
    for (int nt = 0; nt < 4; ++nt) {
      v16h b = *(const v16h*)(pL1 + (((kc * 4 + nt) * 32 + lane) << 4));
      acc[nt] = __builtin_amdgcn_wmma_f32_16x16x32_f16(false, a, false, b,
                                                       (short)0, acc[nt], false, false);
    }
  }

  int nc   = lane & 15;
  int mofs = (lane & 16) ? 8 : 0;
  // bias
#pragma unroll
  for (int nt = 0; nt < 4; ++nt) {
    float bb = b1[nt * 16 + nc];
#pragma unroll
    for (int d = 0; d < 8; ++d) acc[nt][d] += bb;
  }
  // layernorm (per row: 64 values spread over 16 lanes x 4 nt) + relu -> LDS f16
#pragma unroll
  for (int d = 0; d < 8; ++d) {
    float s = acc[0][d] + acc[1][d] + acc[2][d] + acc[3][d];
    float q = acc[0][d]*acc[0][d] + acc[1][d]*acc[1][d] +
              acc[2][d]*acc[2][d] + acc[3][d]*acc[3][d];
    s += __shfl_xor(s, 1, 16); s += __shfl_xor(s, 2, 16);
    s += __shfl_xor(s, 4, 16); s += __shfl_xor(s, 8, 16);
    q += __shfl_xor(q, 1, 16); q += __shfl_xor(q, 2, 16);
    q += __shfl_xor(q, 4, 16); q += __shfl_xor(q, 8, 16);
    float mean = s * (1.0f / 64.0f);
    float var  = q * (1.0f / 64.0f) - mean * mean;
    float rs   = rsqrtf(var + 1e-5f);
    int m = d + mofs;
#pragma unroll
    for (int nt = 0; nt < 4; ++nt) {
      int n = nt * 16 + nc;
      float h = (acc[nt][d] - mean) * rs * ln_g[n] + ln_b[n];
      hbuf[wave][m][n] = (_Float16)fmaxf(h, 0.0f);
    }
  }
  __threadfence_block();

  v8f acc2[4] = {};
#pragma unroll
  for (int kc = 0; kc < 2; ++kc) {
    v16h a = load_a_f16(&hbuf[wave][r][0], kc * 32, lane);
#pragma unroll
    for (int nt = 0; nt < 4; ++nt) {
      v16h b = *(const v16h*)(pL2 + (((kc * 4 + nt) * 32 + lane) << 4));
      acc2[nt] = __builtin_amdgcn_wmma_f32_16x16x32_f16(false, a, false, b,
                                                        (short)0, acc2[nt], false, false);
    }
  }

  long base = (long)tile * 16;
#pragma unroll
  for (int nt = 0; nt < 4; ++nt) {
    int   n  = nt * 16 + nc;
    float bb = b2[n];
#pragma unroll
    for (int d = 0; d < 8; ++d) {
      int m = d + mofs;
      if (m < rowsValid) out[(base + m) * HDIM + n] = acc2[nt][d] + bb;
    }
  }
}

// out = x[N,64] @ W[64,64] + b  (output heads)
__global__ void __launch_bounds__(256) linear64_kernel(
    const float* __restrict__ x, const _Float16* __restrict__ pW,
    const float* __restrict__ b, float* __restrict__ out, int N)
{
  int wave = threadIdx.x >> 5;
  int lane = threadIdx.x & 31;
  int tile = blockIdx.x * WPB + wave;
  if (tile * 16 >= N) return;
  int rowsValid = N - tile * 16;
  if (rowsValid > 16) rowsValid = 16;

  int r = lane & 15;
  long node = (long)tile * 16 + r;
  if (node >= N) node = N - 1;
  const float* row = x + node * HDIM;

  v8f acc[4] = {};
#pragma unroll
  for (int kc = 0; kc < 2; ++kc) {
    v16h a = load_a_f32(row, kc * 32, lane);
#pragma unroll
    for (int nt = 0; nt < 4; ++nt) {
      v16h bf = *(const v16h*)(pW + (((kc * 4 + nt) * 32 + lane) << 4));
      acc[nt] = __builtin_amdgcn_wmma_f32_16x16x32_f16(false, a, false, bf,
                                                       (short)0, acc[nt], false, false);
    }
  }
  int nc   = lane & 15;
  int mofs = (lane & 16) ? 8 : 0;
  long base = (long)tile * 16;
#pragma unroll
  for (int nt = 0; nt < 4; ++nt) {
    int   n  = nt * 16 + nc;
    float bb = b[n];
#pragma unroll
    for (int d = 0; d < 8; ++d) {
      int m = d + mofs;
      if (m < rowsValid) out[(base + m) * HDIM + n] = acc[nt][d] + bb;
    }
  }
}

// ---------------------------------------------------------------------------
extern "C" void kernel_launch(void* const* d_in, const int* in_sizes, int n_in,
                              void* d_out, int out_size, void* d_ws, size_t ws_size,
                              hipStream_t stream) {
  (void)out_size; (void)ws_size;
  const int CF = 5, VF = 3, L = 2;
  const int NC = in_sizes[0] / CF;
  const int NV = in_sizes[1] / VF;
  const int E  = in_sizes[3];

  const float* cons_feat = (const float*)d_in[0];
  const float* var_feat  = (const float*)d_in[1];
  const int*   eidx      = (const int*)d_in[2];   // int32: row0=src, row1=dst
  const float* eattr     = (const float*)d_in[3];
  const int* src = eidx;
  const int* dst = eidx + E;

  // pytree flatten order (dict keys sorted): cons_in{b,w}, cons_out{b,w},
  // layers[ cons_mlp{l1{b,w},l2{b,w},ln_b,ln_g}, cons_msg{l1{b,w},l2{b,w}},
  //         var_mlp{...}, var_msg{...} ], var_in{b,w}, var_out{b,w}
  const float* cons_in_b  = (const float*)d_in[4];
  const float* cons_in_w  = (const float*)d_in[5];
  const float* cons_out_b = (const float*)d_in[6];
  const float* cons_out_w = (const float*)d_in[7];
  int t0 = n_in - 4;
  const float* var_in_b   = (const float*)d_in[t0 + 0];
  const float* var_in_w   = (const float*)d_in[t0 + 1];
  const float* var_out_b  = (const float*)d_in[t0 + 2];
  const float* var_out_w  = (const float*)d_in[t0 + 3];

  struct NodeP { const float *l1b,*l1w,*l2b,*l2w,*lnb,*lng; };
  struct MsgP  { const float *l1b,*l1w,*l2b,*l2w; };
  NodeP cmlp[L], vmlp[L]; MsgP cmsg[L], vmsg[L];
  for (int l = 0; l < L; ++l) {
    int b = 8 + 20 * l;
    cmlp[l] = { (const float*)d_in[b+0], (const float*)d_in[b+1], (const float*)d_in[b+2],
                (const float*)d_in[b+3], (const float*)d_in[b+4], (const float*)d_in[b+5] };
    cmsg[l] = { (const float*)d_in[b+6], (const float*)d_in[b+7],
                (const float*)d_in[b+8], (const float*)d_in[b+9] };
    vmlp[l] = { (const float*)d_in[b+10], (const float*)d_in[b+11], (const float*)d_in[b+12],
                (const float*)d_in[b+13], (const float*)d_in[b+14], (const float*)d_in[b+15] };
    vmsg[l] = { (const float*)d_in[b+16], (const float*)d_in[b+17],
                (const float*)d_in[b+18], (const float*)d_in[b+19] };
  }

  // -------- workspace carve --------
  size_t off = 0;
  auto carve = [&](size_t bytes) -> void* {
    off = (off + 255) & ~(size_t)255;
    void* p = (char*)d_ws + off;
    off += bytes;
    return p;
  };
  int NM = NC > NV ? NC : NV;
  float* nb[4];
  for (int i = 0; i < 4; ++i) nb[i] = (float*)carve((size_t)NM * HDIM * 4);
  float* sum = (float*)carve((size_t)NM * HDIM * 4);
  float* cnt = (float*)carve((size_t)NM * 4);
  const size_t PK128 = (size_t)4 * 4 * 32 * 16 * 2;  // 16 KB
  const size_t PK64  = (size_t)2 * 4 * 32 * 16 * 2;  //  8 KB
  _Float16 *p_cm1[L], *p_cm2[L], *p_cn1[L], *p_cn2[L];
  _Float16 *p_vm1[L], *p_vm2[L], *p_vn1[L], *p_vn2[L];
  for (int l = 0; l < L; ++l) {
    p_cm1[l] = (_Float16*)carve(PK128); p_cm2[l] = (_Float16*)carve(PK64);
    p_cn1[l] = (_Float16*)carve(PK128); p_cn2[l] = (_Float16*)carve(PK64);
    p_vm1[l] = (_Float16*)carve(PK128); p_vm2[l] = (_Float16*)carve(PK64);
    p_vn1[l] = (_Float16*)carve(PK128); p_vn2[l] = (_Float16*)carve(PK64);
  }
  _Float16* p_oc = (_Float16*)carve(PK64);
  _Float16* p_ov = (_Float16*)carve(PK64);

  // -------- pack all weights to f16 B-fragments --------
  for (int l = 0; l < L; ++l) {
    pack_w_kernel<<<4, 128, 0, stream>>>(cmsg[l].l1w, p_cm1[l]);  // rows 0..127 of 129
    pack_w_kernel<<<2, 128, 0, stream>>>(cmsg[l].l2w, p_cm2[l]);
    pack_w_kernel<<<4, 128, 0, stream>>>(cmlp[l].l1w, p_cn1[l]);
    pack_w_kernel<<<2, 128, 0, stream>>>(cmlp[l].l2w, p_cn2[l]);
    pack_w_kernel<<<4, 128, 0, stream>>>(vmsg[l].l1w, p_vm1[l]);
    pack_w_kernel<<<2, 128, 0, stream>>>(vmsg[l].l2w, p_vm2[l]);
    pack_w_kernel<<<4, 128, 0, stream>>>(vmlp[l].l1w, p_vn1[l]);
    pack_w_kernel<<<2, 128, 0, stream>>>(vmlp[l].l2w, p_vn2[l]);
  }
  pack_w_kernel<<<2, 128, 0, stream>>>(cons_out_w, p_oc);
  pack_w_kernel<<<2, 128, 0, stream>>>(var_out_w,  p_ov);

  // -------- input projections --------
  float *chC = nb[0], *vhC = nb[1], *chN = nb[2], *vhN = nb[3];
  long totC = (long)NC * HDIM, totV = (long)NV * HDIM;
  input_proj_kernel<<<(int)((totC + 255) / 256), 256, 0, stream>>>(cons_feat, CF, cons_in_w, cons_in_b, chC, totC);
  input_proj_kernel<<<(int)((totV + 255) / 256), 256, 0, stream>>>(var_feat,  VF, var_in_w,  var_in_b,  vhC, totV);

  int eTiles = (E + 15) / 16;
  int eBlocks = (eTiles + WPB - 1) / WPB;
  int cBlocks = (((NC + 15) / 16) + WPB - 1) / WPB;
  int vBlocks = (((NV + 15) / 16) + WPB - 1) / WPB;

  for (int l = 0; l < L; ++l) {
    // constraint update
    hipMemsetAsync(sum, 0, (size_t)NC * HDIM * 4, stream);
    hipMemsetAsync(cnt, 0, (size_t)NC * 4, stream);
    edge_msg_kernel<<<eBlocks, 256, 0, stream>>>(chC, vhC, dst, src, eattr, E,
        p_cm1[l], cmsg[l].l1w + 128 * HDIM, cmsg[l].l1b, p_cm2[l], cmsg[l].l2b, sum, cnt);
    scatter_fin_kernel<<<(int)((totC + 255) / 256), 256, 0, stream>>>(sum, cnt, totC);
    node_mlp_kernel<<<cBlocks, 256, 0, stream>>>(chC, sum, p_cn1[l], cmlp[l].l1b,
        cmlp[l].lng, cmlp[l].lnb, p_cn2[l], cmlp[l].l2b, chN, NC);

    // variable update (uses new_ch)
    hipMemsetAsync(sum, 0, (size_t)NV * HDIM * 4, stream);
    hipMemsetAsync(cnt, 0, (size_t)NV * 4, stream);
    edge_msg_kernel<<<eBlocks, 256, 0, stream>>>(vhC, chN, src, dst, eattr, E,
        p_vm1[l], vmsg[l].l1w + 128 * HDIM, vmsg[l].l1b, p_vm2[l], vmsg[l].l2b, sum, cnt);
    scatter_fin_kernel<<<(int)((totV + 255) / 256), 256, 0, stream>>>(sum, cnt, totV);
    node_mlp_kernel<<<vBlocks, 256, 0, stream>>>(vhC, sum, p_vn1[l], vmlp[l].l1b,
        vmlp[l].lng, vmlp[l].lnb, p_vn2[l], vmlp[l].l2b, vhN, NV);

    float* t;
    t = chC; chC = chN; chN = t;
    t = vhC; vhC = vhN; vhN = t;
  }

  // -------- output heads --------
  float* outp = (float*)d_out;
  linear64_kernel<<<cBlocks, 256, 0, stream>>>(chC, p_oc, cons_out_b, outp, NC);
  linear64_kernel<<<vBlocks, 256, 0, stream>>>(vhC, p_ov, var_out_b, outp + (long)NC * HDIM, NV);
}
